// MultiHeadedAttention_50044958933141
// MI455X (gfx1250) — matrix operand order repaired and checked by reading, not compiled, not run on hardware
//
#include <hip/hip_runtime.h>
#include <hip/hip_bf16.h>
#include <math.h>

typedef __attribute__((ext_vector_type(16))) _Float16 v16h;
typedef __attribute__((ext_vector_type(8)))  _Float16 v8h;
typedef __attribute__((ext_vector_type(8)))  float    v8f;
typedef __attribute__((ext_vector_type(4)))  float    v4f;
typedef __attribute__((ext_vector_type(4)))  int      v4i;

#define N_HEADS 16
#define N_UNITS 1024
#define DK      64
#define BATCH   2
#define SEQ     2048
#define M_TOTAL (BATCH * SEQ)   // 4096

// ---------------- CDNA5 async global->LDS copy (guarded; falls back to sync) ----
#if defined(__gfx1250__) && __has_builtin(__builtin_amdgcn_global_load_async_to_lds_b128) && \
    __has_builtin(__builtin_amdgcn_s_wait_asynccnt)
#define HAVE_ASYNC_LDS 1
#else
#define HAVE_ASYNC_LDS 0
#endif

#if HAVE_ASYNC_LDS
typedef __attribute__((address_space(1))) v4i gas_v4i;   // prints as "int4 __device__*"
typedef __attribute__((address_space(3))) v4i las_v4i;   // prints as "int4 __shared__*"
#endif

__device__ __forceinline__ void copy16_g2lds(const void* g, void* lds) {
#if HAVE_ASYNC_LDS
    // as1 flat == global address; as3 offset == low 32 bits of the flat shared address
    gas_v4i* gp = (gas_v4i*)(uintptr_t)g;
    las_v4i* lp = (las_v4i*)(uint32_t)(uintptr_t)lds;
    __builtin_amdgcn_global_load_async_to_lds_b128(gp, lp, 0, 0);
#else
    *(v8h*)lds = *(const v8h*)g;
#endif
}

__device__ __forceinline__ void copy_wait() {
#if HAVE_ASYNC_LDS
    __builtin_amdgcn_s_wait_asynccnt(0);
#endif
}

// ---------------- WMMA fragment helpers (16x16x32 f16, wave32) ----------------
// A 16x32 (MxK): lane m=L&15, g=L>>4; halves 0..7 = K 8g..8g+7, halves 8..15 = K 16+8g..23+8g
__device__ __forceinline__ v16h load_a_frag(const _Float16* base, int ldh, int lane) {
    int m = lane & 15, g = lane >> 4;
    const _Float16* p = base + m * ldh;
    v8h lo = *(const v8h*)(p + 8 * g);
    v8h hi = *(const v8h*)(p + 16 + 8 * g);
    v16h a;
#pragma unroll
    for (int i = 0; i < 8; ++i) { a[i] = lo[i]; a[8 + i] = hi[i]; }
    return a;
}

// B 32x16 (KxN) from K-major storage Bt[n][k]: lane n=L&15, g=L>>4; halves 0..7 = K 8g..8g+7, halves 8..15 = K 16+8g..23+8g
__device__ __forceinline__ v16h load_b_frag(const _Float16* baseT, int ldh, int lane) {
    int n = lane & 15, g = lane >> 4;
    const _Float16* p = baseT + n * ldh + 8 * g;
    v8h lo = *(const v8h*)(p);
    v8h hi = *(const v8h*)(p + 16);
    v16h b;
#pragma unroll
    for (int i = 0; i < 8; ++i) { b[i] = lo[i]; b[8 + i] = hi[i]; }
    return b;
}

__device__ __forceinline__ v8f wmma_f16(v16h a, v16h b, v8f c) {
    return __builtin_amdgcn_wmma_f32_16x16x32_f16(false, a, false, b, (short)0, c, false, false);
}

// ---------------- Kernel 1: fused per-head QKV projection ----------------
// C[4096 x 1024] = X[4096 x 1024] * W(head-blocked) + bias; output f16 in (B,H,S,64)
__global__ __launch_bounds__(256)
void qkv_proj_kernel(const float* __restrict__ Xq, const float* __restrict__ Xk, const float* __restrict__ Xv,
                     const float* __restrict__ Wq, const float* __restrict__ Wk, const float* __restrict__ Wv,
                     const float* __restrict__ Bq, const float* __restrict__ Bk, const float* __restrict__ Bv,
                     _Float16* __restrict__ Qo, _Float16* __restrict__ Ko, _Float16* __restrict__ Vo)
{
    __shared__ _Float16 As[128 * 32];   // [m][k]
    __shared__ _Float16 Bs[128 * 32];   // [n][k] (transposed)

    const float* X; const float* W; const float* Bias; _Float16* Out;
    if (blockIdx.z == 0)      { X = Xq; W = Wq; Bias = Bq; Out = Qo; }
    else if (blockIdx.z == 1) { X = Xk; W = Wk; Bias = Bk; Out = Ko; }
    else                      { X = Xv; W = Wv; Bias = Bv; Out = Vo; }

    const int t = threadIdx.x;
    const int wave = t >> 5, lane = t & 31;
    const int wm = wave >> 1, wn = wave & 1;      // 4x2 wave grid -> 128x128 block tile
    const int m0 = blockIdx.y * 128;
    const int n0 = blockIdx.x * 128;

    v8f acc[2][4];
#pragma unroll
    for (int i = 0; i < 2; ++i)
#pragma unroll
        for (int j = 0; j < 4; ++j) acc[i][j] = (v8f){};

    for (int k0 = 0; k0 < N_UNITS; k0 += 32) {
        // A tile: 128x32 f32 -> f16 (1024 float4 loads / 256 threads)
#pragma unroll
        for (int i = 0; i < 4; ++i) {
            int idx = t + i * 256;
            int r = idx >> 3, c4 = idx & 7;
            v4f v = *(const v4f*)(X + (size_t)(m0 + r) * N_UNITS + k0 + c4 * 4);
            _Float16* d = &As[r * 32 + c4 * 4];
            d[0] = (_Float16)v[0]; d[1] = (_Float16)v[1];
            d[2] = (_Float16)v[2]; d[3] = (_Float16)v[3];
        }
        // B tile transposed: Bs[n][k] = W[h, d, dk] where c=n0+n (h=c/64, dk=c%64), d=k0+k
#pragma unroll
        for (int i = 0; i < 16; ++i) {
            int idx = t + i * 256;
            int k = idx >> 7;         // 0..31
            int n = idx & 127;        // 0..127
            int c = n0 + n;
            int d = k0 + k;
            float w = W[(size_t)(c >> 6) * (N_UNITS * DK) + (size_t)d * DK + (c & 63)];
            Bs[n * 32 + k] = (_Float16)w;
        }
        __syncthreads();

        v16h bfr[4];
#pragma unroll
        for (int ni = 0; ni < 4; ++ni)
            bfr[ni] = load_b_frag(&Bs[(wn * 64 + ni * 16) * 32], 32, lane);
#pragma unroll
        for (int mi = 0; mi < 2; ++mi) {
            v16h a = load_a_frag(&As[(wm * 32 + mi * 16) * 32], 32, lane);
#pragma unroll
            for (int ni = 0; ni < 4; ++ni)
                acc[mi][ni] = wmma_f16(a, bfr[ni], acc[mi][ni]);
        }
        __syncthreads();
    }

    // epilogue: +bias, f16 store into (B,H,S,64)
    const int nl = lane & 15, g = lane >> 4;
#pragma unroll
    for (int mi = 0; mi < 2; ++mi) {
#pragma unroll
        for (int ni = 0; ni < 4; ++ni) {
            int c = n0 + wn * 64 + ni * 16 + nl;
            float bias = Bias[c];
            int h = c >> 6, dk = c & 63;
#pragma unroll
            for (int j = 0; j < 8; ++j) {
                int mrow = m0 + wm * 32 + mi * 16 + j + 8 * g;
                int b = mrow >> 11, s = mrow & (SEQ - 1);
                float val = acc[mi][ni][j] + bias;
                Out[((size_t)(b * N_HEADS + h) * SEQ + s) * DK + dk] = (_Float16)val;
            }
        }
    }
}

// ---------------- Kernel 2: flash attention over f16 Q/K/V ----------------
// grid: (S/128, H, B); 8 waves, each wave owns 16 query rows.
__global__ __launch_bounds__(256)
void attn_kernel(const _Float16* __restrict__ Qh, const _Float16* __restrict__ Kh,
                 const _Float16* __restrict__ Vh, const float* __restrict__ mask,
                 _Float16* __restrict__ Hb)
{
    __shared__ _Float16 Kt[64 * 64];        // [key][dk]   (natural == B-layout for Q*K^T)
    __shared__ _Float16 Vt[64 * 64];        // [dk][key]   (transposed == B-layout for P*V)
    __shared__ _Float16 Pls[8 * 16 * 64];   // per-wave probs [wave][m][key]

    const int t = threadIdx.x, wave = t >> 5, lane = t & 31;
    const int g = lane >> 4, nl = lane & 15;
    const int b = blockIdx.z, h = blockIdx.y;
    const int q0 = blockIdx.x * 128 + wave * 16;
    const size_t bh = ((size_t)b * N_HEADS + h) * SEQ;

    // Q fragments held in registers for the whole KV sweep
    const _Float16* qbase = Qh + (bh + q0) * DK;
    const v16h qa0 = load_a_frag(qbase, DK, lane);        // dk 0..31
    const v16h qa1 = load_a_frag(qbase + 32, DK, lane);   // dk 32..63

    v8f O[4];
#pragma unroll
    for (int i = 0; i < 4; ++i) O[i] = (v8f){};
    float mrun[8], lrun[8];
#pragma unroll
    for (int j = 0; j < 8; ++j) { mrun[j] = -1e30f; lrun[j] = 0.0f; }

    const float* mbase = mask + (size_t)b * SEQ * SEQ;

    for (int kc = 0; kc < SEQ; kc += 64) {
        const _Float16* ksrc = Kh + (bh + kc) * DK;
        const _Float16* vsrc = Vh + (bh + kc) * DK;
        // K tile: async byte-for-byte copy into LDS (4096 halves, 2x b128 per thread)
#pragma unroll
        for (int i = 0; i < 2; ++i) {
            int idx = (t + i * 256) * 8;
            copy16_g2lds(ksrc + idx, &Kt[idx]);
        }
        // V tile transposed (goes through VGPRs)
#pragma unroll
        for (int i = 0; i < 16; ++i) {
            int idx = t + i * 256;
            int key = idx >> 6, dk = idx & 63;
            Vt[dk * 64 + key] = vsrc[idx];
        }
        // warm L2/WGP$ for the next chunk while we compute this one
        if (kc + 64 < SEQ) {
            __builtin_prefetch(ksrc + 64 * DK + (t * 16), 0, 1);
            __builtin_prefetch(vsrc + 64 * DK + (t * 16), 0, 1);
        }
        copy_wait();
        __syncthreads();

        // scores = Q * K^T  (16 x 64 per wave)
        v8f sc[4];
#pragma unroll
        for (int ni = 0; ni < 4; ++ni) {
            v8f a = (v8f){};
            v16h bb = load_b_frag(&Kt[ni * 16 * 64], DK, lane);         // dk 0..31
            a = wmma_f16(qa0, bb, a);
            bb = load_b_frag(&Kt[ni * 16 * 64 + 32], DK, lane);         // dk 32..63
            a = wmma_f16(qa1, bb, a);
            sc[ni] = a;
        }

        // scale + mask
#pragma unroll
        for (int ni = 0; ni < 4; ++ni) {
            int keyg = kc + ni * 16 + nl;
#pragma unroll
            for (int j = 0; j < 8; ++j) {
                int qr = q0 + j + 8 * g;
                float s = sc[ni][j] * 0.125f;                 // 1/sqrt(64)
                float mv = mbase[(size_t)qr * SEQ + keyg];
                sc[ni][j] = (mv == 0.0f) ? -1e9f : s;
            }
        }

        // online softmax: rows of a C tile live across 16 lanes in VGPR j
#pragma unroll
        for (int j = 0; j < 8; ++j) {
            float v = fmaxf(fmaxf(sc[0][j], sc[1][j]), fmaxf(sc[2][j], sc[3][j]));
            v = fmaxf(v, __shfl_xor(v, 1, 32));
            v = fmaxf(v, __shfl_xor(v, 2, 32));
            v = fmaxf(v, __shfl_xor(v, 4, 32));
            v = fmaxf(v, __shfl_xor(v, 8, 32));
            float mnew = fmaxf(mrun[j], v);
            float alpha = __expf(mrun[j] - mnew);
            mrun[j] = mnew;
            float rs = 0.0f;
#pragma unroll
            for (int ni = 0; ni < 4; ++ni) {
                float p = __expf(sc[ni][j] - mnew);
                Pls[wave * 1024 + (j + 8 * g) * 64 + ni * 16 + nl] = (_Float16)p;
                rs += p;
            }
            rs += __shfl_xor(rs, 1, 32);
            rs += __shfl_xor(rs, 2, 32);
            rs += __shfl_xor(rs, 4, 32);
            rs += __shfl_xor(rs, 8, 32);
            lrun[j] = lrun[j] * alpha + rs;
#pragma unroll
            for (int tt = 0; tt < 4; ++tt) O[tt][j] *= alpha;
        }
        __syncthreads();    // publish P (C-layout -> A-layout via LDS)

        // O += P * V
        const _Float16* Pw = &Pls[wave * 1024];
        v16h pa0 = load_a_frag(Pw, 64, lane);          // keys 0..31
        v16h pa1 = load_a_frag(Pw + 32, 64, lane);     // keys 32..63
#pragma unroll
        for (int tt = 0; tt < 4; ++tt) {
            v16h bb = load_b_frag(&Vt[tt * 16 * 64], 64, lane);
            O[tt] = wmma_f16(pa0, bb, O[tt]);
            bb = load_b_frag(&Vt[tt * 16 * 64 + 32], 64, lane);
            O[tt] = wmma_f16(pa1, bb, O[tt]);
        }
        __syncthreads();    // done reading K/V/P tiles before next chunk
    }

    // epilogue: O /= l, store f16 into (B, S, H*64) for the output GEMM
#pragma unroll
    for (int tt = 0; tt < 4; ++tt) {
#pragma unroll
        for (int j = 0; j < 8; ++j) {
            int s = q0 + j + 8 * g;
            int col = h * DK + tt * 16 + nl;
            float val = O[tt][j] / lrun[j];
            Hb[((size_t)b * SEQ + s) * N_UNITS + col] = (_Float16)val;
        }
    }
}

// ---------------- Kernel 3: output projection ----------------
// Out[4096 x 1024] f32 = Hb[4096 x 1024] f16 * Wo + bo
__global__ __launch_bounds__(256)
void out_proj_kernel(const _Float16* __restrict__ Hb, const float* __restrict__ Wo,
                     const float* __restrict__ bo, float* __restrict__ Out)
{
    __shared__ _Float16 As[128 * 32];
    __shared__ _Float16 Bs[128 * 32];

    const int t = threadIdx.x;
    const int wave = t >> 5, lane = t & 31;
    const int wm = wave >> 1, wn = wave & 1;
    const int m0 = blockIdx.y * 128;
    const int n0 = blockIdx.x * 128;

    v8f acc[2][4];
#pragma unroll
    for (int i = 0; i < 2; ++i)
#pragma unroll
        for (int j = 0; j < 4; ++j) acc[i][j] = (v8f){};

    for (int k0 = 0; k0 < N_UNITS; k0 += 32) {
        // A tile: already f16 -> async copy (512 b128 / 256 threads = 2 each)
#pragma unroll
        for (int i = 0; i < 2; ++i) {
            int idx = t + i * 256;
            int r = idx >> 2, c8 = idx & 3;
            copy16_g2lds(Hb + (size_t)(m0 + r) * N_UNITS + k0 + c8 * 8,
                         &As[r * 32 + c8 * 8]);
        }
        // B tile transposed from row-major Wo (f32 -> f16 through VGPRs)
#pragma unroll
        for (int i = 0; i < 16; ++i) {
            int idx = t + i * 256;
            int k = idx >> 7;
            int n = idx & 127;
            float w = Wo[(size_t)(k0 + k) * N_UNITS + (n0 + n)];
            Bs[n * 32 + k] = (_Float16)w;
        }
        copy_wait();
        __syncthreads();

        v16h bfr[4];
#pragma unroll
        for (int ni = 0; ni < 4; ++ni)
            bfr[ni] = load_b_frag(&Bs[(wn * 64 + ni * 16) * 32], 32, lane);
#pragma unroll
        for (int mi = 0; mi < 2; ++mi) {
            v16h a = load_a_frag(&As[(wm * 32 + mi * 16) * 32], 32, lane);
#pragma unroll
            for (int ni = 0; ni < 4; ++ni)
                acc[mi][ni] = wmma_f16(a, bfr[ni], acc[mi][ni]);
        }
        __syncthreads();
    }

    const int nl = lane & 15, g = lane >> 4;
#pragma unroll
    for (int mi = 0; mi < 2; ++mi) {
#pragma unroll
        for (int ni = 0; ni < 4; ++ni) {
            int c = n0 + wn * 64 + ni * 16 + nl;
            float bias = bo[c];
#pragma unroll
            for (int j = 0; j < 8; ++j) {
                int mrow = m0 + wm * 32 + mi * 16 + j + 8 * g;
                Out[(size_t)mrow * N_UNITS + c] = acc[mi][ni][j] + bias;
            }
        }
    }
}

// ---------------- host launch ----------------
extern "C" void kernel_launch(void* const* d_in, const int* in_sizes, int n_in,
                              void* d_out, int out_size, void* d_ws, size_t ws_size,
                              hipStream_t stream) {
    const float* query = (const float*)d_in[0];
    const float* key   = (const float*)d_in[1];
    const float* value = (const float*)d_in[2];
    const float* mask  = (const float*)d_in[3];
    const float* Wq    = (const float*)d_in[4];
    const float* bq    = (const float*)d_in[5];
    const float* Wk    = (const float*)d_in[6];
    const float* bk    = (const float*)d_in[7];
    const float* Wv    = (const float*)d_in[8];
    const float* bv    = (const float*)d_in[9];
    const float* Wo    = (const float*)d_in[10];
    const float* bo    = (const float*)d_in[11];
    float* out = (float*)d_out;

    _Float16* ws = (_Float16*)d_ws;
    const size_t TEN = (size_t)M_TOTAL * N_UNITS;   // 4,194,304 halves
    _Float16* Qh = ws;
    _Float16* Kh = ws + TEN;
    _Float16* Vh = ws + 2 * TEN;
    _Float16* Hb = ws + 3 * TEN;

    dim3 blk(256);
    dim3 g1(N_UNITS / 128, M_TOTAL / 128, 3);           // 8 x 32 x 3
    qkv_proj_kernel<<<g1, blk, 0, stream>>>(query, key, value, Wq, Wk, Wv,
                                            bq, bk, bv, Qh, Kh, Vh);

    dim3 g2(SEQ / 128, N_HEADS, BATCH);                 // 16 x 16 x 2
    attn_kernel<<<g2, blk, 0, stream>>>(Qh, Kh, Vh, mask, Hb);

    dim3 g3(N_UNITS / 128, M_TOTAL / 128);              // 8 x 32
    out_proj_kernel<<<g3, blk, 0, stream>>>(Hb, Wo, bo, out);
}
